// CausalSelfAttention_29188597743781
// MI455X (gfx1250) — matrix operand order repaired and checked by reading, not compiled, not run on hardware
//
#include <hip/hip_runtime.h>

// ---------------------------------------------------------------------------
// Causal self-attention for MI455X (gfx1250, wave32, WMMA + async-to-LDS).
// B=4, T=2048, C=1024, H=16, HD=64.
// qkv_gemm (bf16 WMMA, 128x256 tiles) -> flash_attn (bf16 WMMA, exp2-domain
// online softmax, K/V staged via GLOBAL_LOAD_ASYNC_TO_LDS) -> proj_gemm.
// ---------------------------------------------------------------------------

typedef __attribute__((ext_vector_type(16))) __bf16 v16bf;
typedef __attribute__((ext_vector_type(8)))  __bf16 v8bf;
typedef __attribute__((ext_vector_type(4)))  __bf16 v4bf;
typedef __attribute__((ext_vector_type(8)))  float  v8f;
typedef __attribute__((ext_vector_type(4)))  float  v4f;

#define WMMA_BF16(a, b, c) \
  __builtin_amdgcn_wmma_f32_16x16x32_bf16(false, (a), false, (b), (short)0, (c), false, false)

#define BB 4
#define TT 2048
#define CC 1024
#define HH 16
#define HD 64

// Build a 16-element bf16 fragment from two contiguous 8-element (16B) chunks.
// A-frag lane holds K = hi*8 + {0..7} (v0..3) and hi*8 + {16..23} (v4..7);
// B-frag lane holds the same K: hi*8 + {0..7} and hi*8 + {16..23}.
static __device__ __forceinline__ v16bf frag2(const __bf16* p0, const __bf16* p1) {
  v8bf a = *(const v8bf*)p0;
  v8bf b = *(const v8bf*)p1;
  v16bf r;
#pragma unroll
  for (int i = 0; i < 8; ++i) { r[i] = a[i]; r[i + 8] = b[i]; }
  return r;
}

// Async copy 16B global -> LDS (ASYNCcnt-tracked, bypasses VGPRs).
static __device__ __forceinline__ void async_b128(const void* lds_dst, const void* gsrc) {
  unsigned lds_off = (unsigned)(uintptr_t)lds_dst;  // addr[31:0] = LDS offset
  asm volatile("global_load_async_to_lds_b128 %0, %1, off"
               :: "v"(lds_off), "v"(gsrc) : "memory");
}
static __device__ __forceinline__ void wait_async0() {
  asm volatile("s_wait_asynccnt 0x0" ::: "memory");
}

// ---------------------------------------------------------------------------
// Kernel 1: qkv = x @ W_qkv + b_qkv -> Q [B,H,T,HD], K [B,H,T,HD],
// V TRANSPOSED [B,H,HD,T] (so flash_attn can async-copy V rows).
// Block tile 128x256, K-step 32. 8 waves in 2x4 grid, wave tile 64x64.
// N-blocks (256) never straddle the Q/K/V boundaries (multiples of 1024),
// so the Q/K/V select is uniform per block: sel = blockIdx.y >> 2.
// ---------------------------------------------------------------------------
__global__ __launch_bounds__(256) void qkv_gemm_kernel(
    const float* __restrict__ X, const float* __restrict__ W,
    const float* __restrict__ bias,
    __bf16* __restrict__ Qb, __bf16* __restrict__ Kb, __bf16* __restrict__ Vb) {
  const int N = 3 * CC;       // 3072
  const int K = CC;           // 1024

  __shared__ __bf16 As[128 * 40];   // A tile row-major, padded stride 40
  __shared__ __bf16 Bt[256 * 40];   // B tile TRANSPOSED: Bt[n][k], stride 40

  const int tid  = threadIdx.x;
  const int lane = tid & 31;
  const int w    = tid >> 5;
  const int hi   = lane >> 4;
  const int lr   = lane & 15;
  const int wm   = w >> 2;          // 0..1
  const int wn   = w & 3;           // 0..3
  const int m0 = blockIdx.x * 128;
  const int n0 = blockIdx.y * 256;

  v8f acc[4][4];
#pragma unroll
  for (int a = 0; a < 4; ++a)
#pragma unroll
    for (int b2 = 0; b2 < 4; ++b2) acc[a][b2] = {};

  const int arow = tid >> 1, ahalf = tid & 1;  // A staging: 16 floats/thread
  const int brow = tid >> 3, bseg  = tid & 7;  // B staging: 32 floats/thread

  for (int k0 = 0; k0 < K; k0 += 32) {
    __syncthreads();
    // ---- stage A: 128x32 fp32 -> bf16, row-major ----
    {
      const float* src = X + (size_t)(m0 + arow) * K + k0 + ahalf * 16;
      __bf16* dst = As + arow * 40 + ahalf * 16;
#pragma unroll
      for (int c = 0; c < 4; ++c) {
        v4f v = *(const v4f*)(src + c * 4);
        v4bf hv;
#pragma unroll
        for (int e = 0; e < 4; ++e) hv[e] = (__bf16)v[e];
        *(v4bf*)(dst + c * 4) = hv;
      }
    }
    // ---- stage B: 32x256 fp32 -> bf16, transposed into Bt[n][k] ----
    {
      const float* src = W + (size_t)(k0 + brow) * N + n0 + bseg * 32;
#pragma unroll
      for (int c = 0; c < 8; ++c) {
        v4f v = *(const v4f*)(src + c * 4);
#pragma unroll
        for (int e = 0; e < 4; ++e)
          Bt[(bseg * 32 + c * 4 + e) * 40 + brow] = (__bf16)v[e];
      }
    }
    __syncthreads();

    // ---- WMMA compute: 16 wmma / k-step ----
    v16bf bf[4];
#pragma unroll
    for (int ni = 0; ni < 4; ++ni) {
      const __bf16* bp = Bt + (wn * 64 + ni * 16 + lr) * 40 + hi * 8;
      bf[ni] = frag2(bp, bp + 16);
    }
#pragma unroll
    for (int mi = 0; mi < 4; ++mi) {
      const __bf16* ap = As + (wm * 64 + mi * 16 + lr) * 40 + hi * 8;
      v16bf af = frag2(ap, ap + 16);
#pragma unroll
      for (int ni = 0; ni < 4; ++ni)
        acc[mi][ni] = WMMA_BF16(af, bf[ni], acc[mi][ni]);
    }
  }

  // ---- epilogue: hoisted bias, uniform Q/K/V select, imm-offset stores ----
  float bv[4];
#pragma unroll
  for (int ni = 0; ni < 4; ++ni) bv[ni] = bias[n0 + wn * 64 + ni * 16 + lr];

  const int sel = blockIdx.y >> 2;            // 0:Q 1:K 2:V (uniform)
  const int nloc0 = (n0 & 1023) + wn * 64;    // column within the 1024 segment

  if (sel == 2) {
    // V transposed: Vb[(bh*HD + d)*T + t]; i advances t -> imm offset i*2B
#pragma unroll
    for (int mi = 0; mi < 4; ++mi) {
      const int mbase = m0 + wm * 64 + mi * 16 + hi * 8;
      const int b = mbase >> 11, t0 = mbase & 2047;
#pragma unroll
      for (int ni = 0; ni < 4; ++ni) {
        const int nl = nloc0 + ni * 16 + lr;
        const int h = nl >> 6, d = nl & 63;
        __bf16* p = Vb + ((size_t)(b * HH + h) * HD + d) * TT + t0;
#pragma unroll
        for (int i = 0; i < 8; ++i) p[i] = (__bf16)(acc[mi][ni][i] + bv[ni]);
      }
    }
  } else {
    __bf16* dst = sel ? Kb : Qb;
    // Row-major: dst[(bh*T + t)*HD + d]; i advances t -> imm offset i*HD*2B
#pragma unroll
    for (int mi = 0; mi < 4; ++mi) {
      const int mbase = m0 + wm * 64 + mi * 16 + hi * 8;
      const int b = mbase >> 11, t0 = mbase & 2047;
#pragma unroll
      for (int ni = 0; ni < 4; ++ni) {
        const int nl = nloc0 + ni * 16 + lr;
        const int h = nl >> 6, d = nl & 63;
        __bf16* p = dst + ((size_t)(b * HH + h) * TT + t0) * HD + d;
#pragma unroll
        for (int i = 0; i < 8; ++i) p[i * HD] = (__bf16)(acc[mi][ni][i] + bv[ni]);
      }
    }
  }
}

// ---------------------------------------------------------------------------
// Kernel 2: flash attention. grid = (T/128, B*H). 256 threads = 8 waves;
// wave w owns q rows [qblk*128 + w*16, +16). K (row-major) and V (already
// transposed in global) are staged with GLOBAL_LOAD_ASYNC_TO_LDS_B128.
// Softmax runs in the exp2 domain: S2 = S * (log2e/sqrt(HD)).
// ---------------------------------------------------------------------------
__global__ __launch_bounds__(256) void flash_attn_kernel(
    const __bf16* __restrict__ Qb, const __bf16* __restrict__ Kb,
    const __bf16* __restrict__ Vb, __bf16* __restrict__ Y) {
  __shared__ __bf16 Ks[64 * 72];      // K block [key][hd], pad 72
  __shared__ __bf16 Vt[64 * 72];      // V block [hd][key], pad 72
  __shared__ __bf16 Ps[8][16 * 72];   // per-wave P tile [row][key], pad 72

  const int tid  = threadIdx.x;
  const int lane = tid & 31;
  const int w    = tid >> 5;
  const int hi   = lane >> 4;
  const int lr   = lane & 15;
  const int qblk = blockIdx.x;        // 0..15
  const int bh   = blockIdx.y;        // 0..63
  const int b    = bh >> 4;
  const int h    = bh & 15;
  const int q0   = qblk * 128 + w * 16;

  const size_t base = (size_t)bh * TT * HD;
  const __bf16* Qp = Qb + base;
  const __bf16* Kp = Kb + base;       // [T][HD]
  const __bf16* Vp = Vb + base;       // [HD][T]

  // Q A-fragments (persist across the whole key loop).
  v16bf aq[2];
#pragma unroll
  for (int kc = 0; kc < 2; ++kc) {
    const __bf16* p = Qp + (size_t)(q0 + lr) * HD + kc * 32 + hi * 8;
    aq[kc] = frag2(p, p + 16);
  }

  float mrow[8], lrow[8];
#pragma unroll
  for (int i = 0; i < 8; ++i) { mrow[i] = -__builtin_inff(); lrow[i] = 0.f; }
  v8f o[4];
#pragma unroll
  for (int n = 0; n < 4; ++n) o[n] = {};

  const float scale2 = 0.18033688011112042f;  // log2(e)/sqrt(HD)
  const int nkb = 2 * qblk + 2;               // causal key-block bound
  const int srow = tid >> 2, sseg = tid & 3;  // staging: 32B per thread/array

  for (int kb = 0; kb < nkb; ++kb) {
    const int kbase = kb * 64;
    __syncthreads();
    // ---- async-stage K (64 rows x 128B) and V (64 hd-rows x 128B) ----
    {
      const __bf16* ks = Kp + (size_t)(kbase + srow) * HD + sseg * 16;
      __bf16* kd = Ks + srow * 72 + sseg * 16;
      async_b128(kd,     ks);
      async_b128(kd + 8, ks + 8);

      const __bf16* vs = Vp + (size_t)srow * TT + kbase + sseg * 16;
      __bf16* vd = Vt + srow * 72 + sseg * 16;
      async_b128(vd,     vs);
      async_b128(vd + 8, vs + 8);
    }
    wait_async0();
    __syncthreads();

    // ---- S = Q @ K^T (4 n-chunks of 16 keys, 2 WMMA each over HD=64) ----
    v8f st[4];
#pragma unroll
    for (int n16 = 0; n16 < 4; ++n16) {
      const __bf16* bp = Ks + (n16 * 16 + lr) * 72 + hi * 8;
      v16bf bk0 = frag2(bp, bp + 16);        // hd 0..31
      v16bf bk1 = frag2(bp + 32, bp + 48);   // hd 32..63
      v8f s = {};
      s = WMMA_BF16(aq[0], bk0, s);
      s = WMMA_BF16(aq[1], bk1, s);
      st[n16] = s;
    }

    // ---- scale (exp2 domain) + causal mask + row max ----
    float rm[8];
#pragma unroll
    for (int i = 0; i < 8; ++i) rm[i] = -__builtin_inff();
#pragma unroll
    for (int n16 = 0; n16 < 4; ++n16) {
      const int col = kbase + n16 * 16 + lr;
#pragma unroll
      for (int i = 0; i < 8; ++i) {
        const int row = q0 + i + hi * 8;
        float f = st[n16][i] * scale2;
        f = (col > row) ? -__builtin_inff() : f;
        st[n16][i] = f;
        rm[i] = fmaxf(rm[i], f);
      }
    }
#pragma unroll
    for (int i = 0; i < 8; ++i) {
#pragma unroll
      for (int off = 8; off >= 1; off >>= 1)
        rm[i] = fmaxf(rm[i], __shfl_xor(rm[i], off, 32));
    }

    float alpha[8], ps[8];
#pragma unroll
    for (int i = 0; i < 8; ++i) {
      const float mn = fmaxf(mrow[i], rm[i]);
      alpha[i] = __builtin_amdgcn_exp2f(mrow[i] - mn);
      mrow[i] = mn;
      ps[i] = 0.f;
    }

    // ---- P = exp2(S2 - m): bf16 into per-wave LDS (D -> A relayout) ----
#pragma unroll
    for (int n16 = 0; n16 < 4; ++n16) {
#pragma unroll
      for (int i = 0; i < 8; ++i) {
        const float p = __builtin_amdgcn_exp2f(st[n16][i] - mrow[i]);
        ps[i] += p;
        Ps[w][(i + hi * 8) * 72 + n16 * 16 + lr] = (__bf16)p;
      }
    }
#pragma unroll
    for (int i = 0; i < 8; ++i) {
#pragma unroll
      for (int off = 8; off >= 1; off >>= 1)
        ps[i] += __shfl_xor(ps[i], off, 32);
      lrow[i] = lrow[i] * alpha[i] + ps[i];
    }

    // ---- rescale O, then O += P @ V ----
#pragma unroll
    for (int n = 0; n < 4; ++n)
#pragma unroll
      for (int i = 0; i < 8; ++i) o[n][i] *= alpha[i];

    v16bf pa[2];
#pragma unroll
    for (int kc = 0; kc < 2; ++kc) {
      const __bf16* p = &Ps[w][lr * 72 + kc * 32 + hi * 8];
      pa[kc] = frag2(p, p + 16);
    }
#pragma unroll
    for (int nh = 0; nh < 4; ++nh) {
      const __bf16* vp = Vt + (nh * 16 + lr) * 72 + hi * 8;
      v16bf bv0 = frag2(vp, vp + 16);        // keys 0..31
      v16bf bv1 = frag2(vp + 32, vp + 48);   // keys 32..63
      o[nh] = WMMA_BF16(pa[0], bv0, o[nh]);
      o[nh] = WMMA_BF16(pa[1], bv1, o[nh]);
    }
  }

  // ---- epilogue: O /= l, single base + imm-offset bf16 stores ----
  float linv[8];
#pragma unroll
  for (int i = 0; i < 8; ++i) linv[i] = 1.f / lrow[i];
  __bf16* Yp = Y + ((size_t)(b * TT + q0 + hi * 8)) * CC + h * HD + lr;
#pragma unroll
  for (int nh = 0; nh < 4; ++nh)
#pragma unroll
    for (int i = 0; i < 8; ++i)
      Yp[(size_t)i * CC + nh * 16] = (__bf16)(o[nh][i] * linv[i]);
}

// ---------------------------------------------------------------------------
// Kernel 3: out = Yatt @ W_proj + b_proj (fp32 to d_out).
// Block tile 128x256, wave tile 64x64; bf16 A staged via async-to-LDS.
// ---------------------------------------------------------------------------
__global__ __launch_bounds__(256) void proj_gemm_kernel(
    const __bf16* __restrict__ A, const float* __restrict__ W,
    const float* __restrict__ bias, float* __restrict__ Out) {
  const int N = CC, K = CC;

  __shared__ __bf16 As[128 * 40];
  __shared__ __bf16 Bt[256 * 40];

  const int tid  = threadIdx.x;
  const int lane = tid & 31;
  const int w    = tid >> 5;
  const int hi   = lane >> 4;
  const int lr   = lane & 15;
  const int wm   = w >> 2;
  const int wn   = w & 3;
  const int m0 = blockIdx.x * 128;
  const int n0 = blockIdx.y * 256;

  v8f acc[4][4];
#pragma unroll
  for (int a = 0; a < 4; ++a)
#pragma unroll
    for (int b2 = 0; b2 < 4; ++b2) acc[a][b2] = {};

  const int arow = tid >> 1, ahalf = tid & 1;
  const int brow = tid >> 3, bseg  = tid & 7;

  for (int k0 = 0; k0 < K; k0 += 32) {
    __syncthreads();
    {  // A already bf16: async copy straight into LDS
      const __bf16* src = A + (size_t)(m0 + arow) * K + k0 + ahalf * 16;
      __bf16* dst = As + arow * 40 + ahalf * 16;
      async_b128(dst,     src);
      async_b128(dst + 8, src + 8);
    }
    {  // B fp32 -> bf16 transposed
      const float* src = W + (size_t)(k0 + brow) * N + n0 + bseg * 32;
#pragma unroll
      for (int c = 0; c < 8; ++c) {
        v4f v = *(const v4f*)(src + c * 4);
#pragma unroll
        for (int e = 0; e < 4; ++e)
          Bt[(bseg * 32 + c * 4 + e) * 40 + brow] = (__bf16)v[e];
      }
    }
    wait_async0();
    __syncthreads();

    v16bf bf[4];
#pragma unroll
    for (int ni = 0; ni < 4; ++ni) {
      const __bf16* bp = Bt + (wn * 64 + ni * 16 + lr) * 40 + hi * 8;
      bf[ni] = frag2(bp, bp + 16);
    }
#pragma unroll
    for (int mi = 0; mi < 4; ++mi) {
      const __bf16* ap = As + (wm * 64 + mi * 16 + lr) * 40 + hi * 8;
      v16bf af = frag2(ap, ap + 16);
#pragma unroll
      for (int ni = 0; ni < 4; ++ni)
        acc[mi][ni] = WMMA_BF16(af, bf[ni], acc[mi][ni]);
    }
  }

  // ---- epilogue: hoisted bias, base + imm-offset fp32 stores ----
  float bv[4];
#pragma unroll
  for (int ni = 0; ni < 4; ++ni) bv[ni] = bias[n0 + wn * 64 + ni * 16 + lr];
#pragma unroll
  for (int mi = 0; mi < 4; ++mi) {
    const int mbase = m0 + wm * 64 + mi * 16 + hi * 8;
#pragma unroll
    for (int ni = 0; ni < 4; ++ni) {
      float* p = Out + (size_t)mbase * N + n0 + wn * 64 + ni * 16 + lr;
#pragma unroll
      for (int i = 0; i < 8; ++i) p[i * N] = acc[mi][ni][i] + bv[ni];
    }
  }
}

// ---------------------------------------------------------------------------
extern "C" void kernel_launch(void* const* d_in, const int* in_sizes, int n_in,
                              void* d_out, int out_size, void* d_ws, size_t ws_size,
                              hipStream_t stream) {
  const float* x     = (const float*)d_in[0];
  const float* Wqkv  = (const float*)d_in[1];
  const float* bqkv  = (const float*)d_in[2];
  const float* Wproj = (const float*)d_in[3];
  const float* bproj = (const float*)d_in[4];
  float* out = (float*)d_out;

  const size_t NE = (size_t)BB * HH * TT * HD;  // 8M elements per Q/K/V buffer
  __bf16* Qb = (__bf16*)d_ws;
  __bf16* Kb = Qb + NE;
  __bf16* Vb = Kb + NE;                          // stored [B,H,HD,T]
  __bf16* Ya = Vb + NE;                          // [B,T,C] bf16

  // qkv: M=8192, N=3072 -> grid (64, 12)
  qkv_gemm_kernel<<<dim3(64, 12), 256, 0, stream>>>(x, Wqkv, bqkv, Qb, Kb, Vb);
  // attention: grid (T/128 = 16, B*H = 64)
  flash_attn_kernel<<<dim3(16, 64), 256, 0, stream>>>(Qb, Kb, Vb, Ya);
  // projection: M=8192, N=1024 -> grid (64, 4)
  proj_gemm_kernel<<<dim3(64, 4), 256, 0, stream>>>(Ya, Wproj, bproj, out);

  (void)in_sizes; (void)n_in; (void)out_size; (void)ws_size;
}
